// GAT_10307921510704
// MI455X (gfx1250) — compile-verified, matrix operand order checked by reading, not run on hardware
//
#include <hip/hip_runtime.h>
#include <hip/hip_bf16.h>
#include <math.h>

// ---------------- problem constants (from reference) ----------------
#define NN      50000        // nodes
#define EE      1600000      // edges (before self loops)
#define ETOT    (EE + NN)    // edges incl. self loops
#define FIN     128          // input features (= H*C1 too)
#define HEADS   4
#define C1      32           // hidden per head
#define C2      32           // F_OUT

typedef __attribute__((ext_vector_type(16))) _Float16 half16_t;
typedef __attribute__((ext_vector_type(8)))  _Float16 half8_t;
typedef __attribute__((ext_vector_type(8)))  float    float8_t;

// ---------------- helpers ----------------
__device__ __forceinline__ unsigned fkey(float f) {
    unsigned u = __float_as_uint(f);
    return (u & 0x80000000u) ? ~u : (u | 0x80000000u);
}
__device__ __forceinline__ float fdec(unsigned k) {
    return (k & 0x80000000u) ? __uint_as_float(k ^ 0x80000000u) : __uint_as_float(~k);
}
// fkey(-inf) = ~0xFF800000
#define NEG_INF_KEY 0x007FFFFFu

__device__ __forceinline__ float lrelu02(float v) { return v > 0.f ? v : 0.2f * v; }

// ---------------- fills ----------------
__global__ void fill_f32(float* __restrict__ p, float v, int n) {
    int i = blockIdx.x * blockDim.x + threadIdx.x;
    if (i < n) p[i] = v;
}
__global__ void fill_u32(unsigned* __restrict__ p, unsigned v, int n) {
    int i = blockIdx.x * blockDim.x + threadIdx.x;
    if (i < n) p[i] = v;
}

// ---------------- f32 -> f16 conversions (GEMM operands) ----------------
__global__ void cvt_x_f16(const float* __restrict__ x, _Float16* __restrict__ xh, int n) {
    int i = blockIdx.x * blockDim.x + threadIdx.x;
    if (i < n) xh[i] = (_Float16)x[i];
}
// W1 [K=128][N=128] row-major -> Bt [n][k] f16
__global__ void cvt_w1t(const float* __restrict__ W, _Float16* __restrict__ Bt) {
    int i = blockIdx.x * blockDim.x + threadIdx.x;
    if (i >= 128 * 128) return;
    int k = i >> 7, n = i & 127;
    Bt[n * 128 + k] = (_Float16)W[i];
}
// W2 [K=128][N=32] row-major -> Bt [n][k] f16
__global__ void cvt_w2t(const float* __restrict__ W, _Float16* __restrict__ Bt) {
    int i = blockIdx.x * blockDim.x + threadIdx.x;
    if (i >= 128 * 32) return;
    int k = i >> 5, n = i & 31;
    Bt[n * 128 + k] = (_Float16)W[i];
}

// ---------------- WMMA GEMM: C[Nrows x NT*16] = A[Nrows x 128] * Bt^T ----------------
// One wave32 per 16-row strip; the A fragment is loaded once per K-step and
// reused across all NT column tiles (v_wmma_f32_16x16x32_f16, f32 accumulate).
template <int NT>
__global__ __launch_bounds__(256)
void gemm_f16_wmma(const _Float16* __restrict__ A, const _Float16* __restrict__ Bt,
                   float* __restrict__ C, int nRowTiles) {
    const int wave = threadIdx.x >> 5;
    const int lane = threadIdx.x & 31;
    const int rowTile = blockIdx.x * 8 + wave;
    if (rowTile >= nRowTiles) return;           // wave-uniform exit

    const int r   = lane & 15;
    const int hi  = lane >> 4;
    const int m0  = rowTile * 16;
    const int ldc = NT * 16;

    float8_t acc[NT];
    float8_t zero = {};
#pragma unroll
    for (int t = 0; t < NT; ++t) acc[t] = zero;

    const _Float16* arow = A + (size_t)(m0 + r) * 128;

#pragma unroll
    for (int ks = 0; ks < 4; ++ks) {
        const int kb = ks * 32 + hi * 8;        // ISA 16-bit A layout: lane-half K split
        half8_t alo = *(const half8_t*)(arow + kb);
        half8_t ahi = *(const half8_t*)(arow + kb + 16);
        half16_t afrag;
#pragma unroll
        for (int i = 0; i < 8; ++i) { afrag[i] = alo[i]; afrag[8 + i] = ahi[i]; }

#pragma unroll
        for (int t = 0; t < NT; ++t) {
            const _Float16* brow = Bt + (size_t)(t * 16 + r) * 128;
            half8_t blo = *(const half8_t*)(brow + kb);
            half8_t bhi = *(const half8_t*)(brow + kb + 16);
            half16_t bfrag;
#pragma unroll
            for (int i = 0; i < 8; ++i) { bfrag[i] = blo[i]; bfrag[8 + i] = bhi[i]; }
            acc[t] = __builtin_amdgcn_wmma_f32_16x16x32_f16(
                false, afrag, false, bfrag, (short)0, acc[t], false, false);
        }
    }

    // f32 C/D layout: VGPR i -> row m0 + hi*8 + i, col = lane (within 16)
#pragma unroll
    for (int t = 0; t < NT; ++t) {
#pragma unroll
        for (int i = 0; i < 8; ++i)
            C[(size_t)(m0 + hi * 8 + i) * ldc + t * 16 + r] = acc[t][i];
    }
}

// ---------------- attention logits ----------------
__global__ void alpha1_kernel(const float* __restrict__ xw,
                              const float* __restrict__ att_s, const float* __restrict__ att_d,
                              float* __restrict__ asrc, float* __restrict__ adst) {
    int t = blockIdx.x * blockDim.x + threadIdx.x;
    if (t >= NN * HEADS) return;
    int n = t >> 2, h = t & 3;
    const float* row = xw + (size_t)n * FIN + h * C1;
    float ss = 0.f, sd = 0.f;
#pragma unroll
    for (int c = 0; c < C1; ++c) {
        float v = row[c];
        ss += v * att_s[h * C1 + c];
        sd += v * att_d[h * C1 + c];
    }
    asrc[t] = ss; adst[t] = sd;
}
__global__ void alpha2_kernel(const float* __restrict__ xw,
                              const float* __restrict__ att_s, const float* __restrict__ att_d,
                              float* __restrict__ asrc, float* __restrict__ adst) {
    int n = blockIdx.x * blockDim.x + threadIdx.x;
    if (n >= NN) return;
    const float* row = xw + (size_t)n * C2;
    float ss = 0.f, sd = 0.f;
#pragma unroll
    for (int c = 0; c < C2; ++c) { ss += row[c] * att_s[c]; sd += row[c] * att_d[c]; }
    asrc[n] = ss; adst[n] = sd;
}

// ---------------- edge helpers ----------------
__device__ __forceinline__ void edge_sd(const long long* __restrict__ ei, int e, int& s, int& d) {
    if (e < EE) { s = (int)ei[e]; d = (int)ei[EE + e]; }
    else        { s = d = e - EE; }
}

// ---------------- layer 1 edge passes ----------------
__global__ void edge_max1(const long long* __restrict__ ei,
                          const float* __restrict__ asrc, const float* __restrict__ adst,
                          unsigned* __restrict__ emax) {
    int e = blockIdx.x * blockDim.x + threadIdx.x;
    if (e >= ETOT) return;
    __builtin_prefetch(ei + e + 8192, 0, 0);    // stream edge list (global_prefetch_b8)
    int s, d; edge_sd(ei, e, s, d);
#pragma unroll
    for (int h = 0; h < HEADS; ++h) {
        float v = lrelu02(asrc[s * HEADS + h] + adst[d * HEADS + h]);
        atomicMax(&emax[d * HEADS + h], fkey(v));
    }
}
__global__ void edge_sum1(const long long* __restrict__ ei,
                          const float* __restrict__ asrc, const float* __restrict__ adst,
                          const unsigned* __restrict__ emax, float* __restrict__ esum) {
    int e = blockIdx.x * blockDim.x + threadIdx.x;
    if (e >= ETOT) return;
    int s, d; edge_sd(ei, e, s, d);
#pragma unroll
    for (int h = 0; h < HEADS; ++h) {
        float v  = lrelu02(asrc[s * HEADS + h] + adst[d * HEADS + h]);
        float ex = expf(v - fdec(emax[d * HEADS + h]));
        atomicAdd(&esum[d * HEADS + h], ex);
    }
}
// one wave32 per edge; lane = channel; 4 heads -> 128 floats gathered+scattered
__global__ __launch_bounds__(256)
void edge_scatter1(const long long* __restrict__ ei,
                   const float* __restrict__ asrc, const float* __restrict__ adst,
                   const unsigned* __restrict__ emax, const float* __restrict__ esum,
                   const float* __restrict__ xw, float* __restrict__ out) {
    int e = blockIdx.x * 8 + (threadIdx.x >> 5);
    if (e >= ETOT) return;                      // wave-uniform
    int lane = threadIdx.x & 31;
    int s, d; edge_sd(ei, e, s, d);
#pragma unroll
    for (int h = 0; h < HEADS; ++h) {
        float v     = lrelu02(asrc[s * HEADS + h] + adst[d * HEADS + h]);
        float ex    = expf(v - fdec(emax[d * HEADS + h]));
        float alpha = ex / (esum[d * HEADS + h] + 1e-16f);
        float contrib = xw[(size_t)s * FIN + h * C1 + lane] * alpha;
        atomicAdd(&out[(size_t)d * FIN + h * C1 + lane], contrib);
    }
}

// ---------------- bias + ELU -> f16 operand for layer-2 GEMM ----------------
__global__ void bias_elu(const float* __restrict__ agg, const float* __restrict__ b,
                         _Float16* __restrict__ hh) {
    int t = blockIdx.x * blockDim.x + threadIdx.x;
    if (t >= NN * FIN) return;
    float v = agg[t] + b[t & 127];
    float h = v > 0.f ? v : (expf(v) - 1.f);
    hh[t] = (_Float16)h;
}

// ---------------- layer 2 edge passes (1 head, 32 channels) ----------------
__global__ void edge_max2(const long long* __restrict__ ei,
                          const float* __restrict__ asrc, const float* __restrict__ adst,
                          unsigned* __restrict__ emax) {
    int e = blockIdx.x * blockDim.x + threadIdx.x;
    if (e >= ETOT) return;
    int s, d; edge_sd(ei, e, s, d);
    float v = lrelu02(asrc[s] + adst[d]);
    atomicMax(&emax[d], fkey(v));
}
__global__ void edge_sum2(const long long* __restrict__ ei,
                          const float* __restrict__ asrc, const float* __restrict__ adst,
                          const unsigned* __restrict__ emax, float* __restrict__ esum) {
    int e = blockIdx.x * blockDim.x + threadIdx.x;
    if (e >= ETOT) return;
    int s, d; edge_sd(ei, e, s, d);
    float v = lrelu02(asrc[s] + adst[d]);
    atomicAdd(&esum[d], expf(v - fdec(emax[d])));
}
__global__ __launch_bounds__(256)
void edge_scatter2(const long long* __restrict__ ei,
                   const float* __restrict__ asrc, const float* __restrict__ adst,
                   const unsigned* __restrict__ emax, const float* __restrict__ esum,
                   const float* __restrict__ xw, float* __restrict__ out) {
    int e = blockIdx.x * 8 + (threadIdx.x >> 5);
    if (e >= ETOT) return;
    int lane = threadIdx.x & 31;
    int s, d; edge_sd(ei, e, s, d);
    float v     = lrelu02(asrc[s] + adst[d]);
    float ex    = expf(v - fdec(emax[d]));
    float alpha = ex / (esum[d] + 1e-16f);
    atomicAdd(&out[(size_t)d * C2 + lane], xw[(size_t)s * C2 + lane] * alpha);
}

// ---------------- final: +b2 and L2 row-normalize (in place on d_out) ----------------
__global__ void finalize(float* __restrict__ z, const float* __restrict__ b2) {
    int n = blockIdx.x * blockDim.x + threadIdx.x;
    if (n >= NN) return;
    float v[C2]; float ss = 0.f;
#pragma unroll
    for (int c = 0; c < C2; ++c) { v[c] = z[(size_t)n * C2 + c] + b2[c]; ss += v[c] * v[c]; }
    float nrm = sqrtf(ss);
#pragma unroll
    for (int c = 0; c < C2; ++c) z[(size_t)n * C2 + c] = v[c] / nrm;
}

// ---------------- host side ----------------
extern "C" void kernel_launch(void* const* d_in, const int* in_sizes, int n_in,
                              void* d_out, int out_size, void* d_ws, size_t ws_size,
                              hipStream_t stream) {
    const float*     x   = (const float*)d_in[0];
    const long long* ei  = (const long long*)d_in[1];   // reference dtype: int64
    const float*     W1  = (const float*)d_in[2];
    const float*     as1 = (const float*)d_in[3];
    const float*     ad1 = (const float*)d_in[4];
    const float*     b1  = (const float*)d_in[5];
    const float*     W2  = (const float*)d_in[6];
    const float*     as2 = (const float*)d_in[7];
    const float*     ad2 = (const float*)d_in[8];
    const float*     b2  = (const float*)d_in[9];
    float* out = (float*)d_out;

    // carve workspace (all L2-resident: ~86 MB total)
    char* ws = (char*)d_ws;
    size_t off = 0;
    auto carve = [&](size_t bytes) -> void* {
        void* p = ws + off; off += (bytes + 255) & ~(size_t)255; return p;
    };
    _Float16* xh    = (_Float16*)carve((size_t)NN * FIN * 2);
    _Float16* hh    = (_Float16*)carve((size_t)NN * FIN * 2);
    _Float16* w1t   = (_Float16*)carve((size_t)128 * 128 * 2);
    _Float16* w2t   = (_Float16*)carve((size_t)C2 * 128 * 2);
    float*    xw1   = (float*)carve((size_t)NN * FIN * 4);
    float*    out1  = (float*)carve((size_t)NN * FIN * 4);
    float*    xw2   = (float*)carve((size_t)NN * C2 * 4);
    float*    asrc1 = (float*)carve((size_t)NN * HEADS * 4);
    float*    adst1 = (float*)carve((size_t)NN * HEADS * 4);
    unsigned* emax1 = (unsigned*)carve((size_t)NN * HEADS * 4);
    float*    esum1 = (float*)carve((size_t)NN * HEADS * 4);
    float*    asrc2 = (float*)carve((size_t)NN * 4);
    float*    adst2 = (float*)carve((size_t)NN * 4);
    unsigned* emax2 = (unsigned*)carve((size_t)NN * 4);
    float*    esum2 = (float*)carve((size_t)NN * 4);

    const int B = 256;
    auto blk = [](long long n, int b) { return (int)((n + b - 1) / b); };
    const int rowTiles = NN / 16;              // 50000/16 = 3125 exactly
    const int gemmBlocks = blk(rowTiles, 8);
    const int edgeWaveBlocks = blk(ETOT, 8);

    // ---- prep: f16 operands ----
    cvt_x_f16<<<blk((long long)NN * FIN, B), B, 0, stream>>>(x, xh, NN * FIN);
    cvt_w1t<<<blk(128 * 128, B), B, 0, stream>>>(W1, w1t);
    cvt_w2t<<<blk(128 * 32, B), B, 0, stream>>>(W2, w2t);

    // ---- layer 1 ----
    fill_f32<<<blk((long long)NN * FIN, B), B, 0, stream>>>(out1, 0.f, NN * FIN);
    fill_f32<<<blk(NN * HEADS, B), B, 0, stream>>>(esum1, 0.f, NN * HEADS);
    fill_u32<<<blk(NN * HEADS, B), B, 0, stream>>>(emax1, NEG_INF_KEY, NN * HEADS);

    gemm_f16_wmma<8><<<gemmBlocks, B, 0, stream>>>(xh, w1t, xw1, rowTiles);
    alpha1_kernel<<<blk(NN * HEADS, B), B, 0, stream>>>(xw1, as1, ad1, asrc1, adst1);

    edge_max1<<<blk(ETOT, B), B, 0, stream>>>(ei, asrc1, adst1, emax1);
    edge_sum1<<<blk(ETOT, B), B, 0, stream>>>(ei, asrc1, adst1, emax1, esum1);
    edge_scatter1<<<edgeWaveBlocks, B, 0, stream>>>(ei, asrc1, adst1, emax1, esum1, xw1, out1);

    bias_elu<<<blk((long long)NN * FIN, B), B, 0, stream>>>(out1, b1, hh);

    // ---- layer 2 ----
    fill_f32<<<blk((long long)NN * C2, B), B, 0, stream>>>(out, 0.f, NN * C2);
    fill_f32<<<blk(NN, B), B, 0, stream>>>(esum2, 0.f, NN);
    fill_u32<<<blk(NN, B), B, 0, stream>>>(emax2, NEG_INF_KEY, NN);

    gemm_f16_wmma<2><<<gemmBlocks, B, 0, stream>>>(hh, w2t, xw2, rowTiles);
    alpha2_kernel<<<blk(NN, B), B, 0, stream>>>(xw2, as2, ad2, asrc2, adst2);

    edge_max2<<<blk(ETOT, B), B, 0, stream>>>(ei, asrc2, adst2, emax2);
    edge_sum2<<<blk(ETOT, B), B, 0, stream>>>(ei, asrc2, adst2, emax2, esum2);
    edge_scatter2<<<edgeWaveBlocks, B, 0, stream>>>(ei, asrc2, adst2, emax2, esum2, xw2, out);

    finalize<<<blk(NN, B), B, 0, stream>>>(out, b2);
}
